// LSTMDecoder_2156073583148
// MI455X (gfx1250) — matrix operand order repaired and checked by reading, not compiled, not run on hardware
//
#include <hip/hip_runtime.h>
#include <hip/hip_bf16.h>
#include <math.h>
#include <stdint.h>

// LSTM decoder with attention for MI455X (gfx1250, wave32, WMMA).
// I=256, H=512, O=256, B=128, S_dec=S_enc=256.
// All matrix operands pre-converted to f16 once; inner loops are
// global_load_async_to_lds_b128 -> ds_load_b128 -> v_wmma_f32_16x16x32_f16.
// K/V (fp32, 128 MB) + f16 operands stay resident in the 192 MB L2.

typedef __attribute__((ext_vector_type(16))) _Float16 v16h;
typedef __attribute__((ext_vector_type(8)))  _Float16 v8h;
typedef __attribute__((ext_vector_type(8)))  float    v8f;

#define I_DIM 256
#define H_DIM 512
#define O_DIM 256
#define B_DIM 128
#define SDEC  256
#define SENC  256

__device__ __forceinline__ float sigmoidf_(float x) { return 1.0f / (1.0f + __expf(-x)); }

// ---- async staging: copy 16 rows x ldHalves f16 from global into LDS -------
// Each lane moves 16B per instruction (GLOBAL_LOAD_ASYNC_TO_LDS_B128, GV mode).
// LDS byte address = low 32 bits of the generic shared pointer (LDS aperture).
__device__ __forceinline__ void stage_rows_async(
    _Float16* dst, const _Float16* src, int ldHalves, size_t rowStrideHalves, int tid)
{
    const int cpr = ldHalves >> 3;                 // 16-byte chunks per row
    for (int idx = tid; idx < 16 * cpr; idx += 128) {
        const int r = idx / cpr, qq = idx - r * cpr;
        uint32_t lds_addr = (uint32_t)(uintptr_t)(dst + r * ldHalves + 8 * qq);
        uint64_t gaddr    = (uint64_t)(uintptr_t)(src + (size_t)r * rowStrideHalves + 8 * qq);
        asm volatile("global_load_async_to_lds_b128 %0, %1, off"
                     :: "v"(lds_addr), "v"(gaddr) : "memory");
    }
}
__device__ __forceinline__ void wait_async_and_barrier() {
    asm volatile("s_wait_asynccnt 0x0" ::: "memory");
    __syncthreads();
}

// A-fragment (16x32 f16) from f16 LDS block, row stride ld (halves).
// lane<16 -> M=lane, K 0..15; lane>=16 -> M=lane-16, K 16..31.
__device__ __forceinline__ v16h load_a_frag16(const _Float16* base, int ld, int kk, int lane) {
    const int r  = lane & 15;
    const int kb = (lane < 16) ? 0 : 16;
    const _Float16* p = base + r * ld + kk + kb;
    v8h lo = *(const v8h*)(p);
    v8h hi = *(const v8h*)(p + 8);
    return __builtin_shufflevector(lo, hi, 0,1,2,3,4,5,6,7,8,9,10,11,12,13,14,15);
}

// B-fragment (32x16 f16): column n = contiguous f16 weight row (A @ W^T).
// lane<16 -> N=lane, K 0..15; lane>=16 -> N=lane-16, K 16..31 (one 32B load).
__device__ __forceinline__ v16h load_b_frag16(const _Float16* wrow, int kk, int lane) {
    const int kb = (lane < 16) ? 0 : 16;
    return *(const v16h*)(wrow + kk + kb);
}

// ---- generic C[M,N] = A16[M,512] @ Bw16[N,512]^T + bias; optional f16 copy --
// grid = (M/16, N/64), block = 128 (4 waves, one 16x16 tile each).
__global__ __launch_bounds__(128) void gemm_bias_wmma_f16(
    const _Float16* __restrict__ A16, int lda,
    const _Float16* __restrict__ Bw16, int ldb,
    const float* __restrict__ bias,
    float* __restrict__ C, int ldc,
    _Float16* __restrict__ C16)
{
    constexpr int KD = 512;
    __shared__ __align__(32) _Float16 As[16 * KD];      // 16 KB
    const int tid = threadIdx.x;
    const int m0  = blockIdx.x * 16;

    stage_rows_async(As, A16 + (size_t)m0 * lda, KD, lda, tid);
    wait_async_and_barrier();

    const int wave = tid >> 5, lane = tid & 31;
    const int n0 = blockIdx.y * 64 + wave * 16;
    const int nB = lane & 15;
    const _Float16* Brow = Bw16 + (size_t)(n0 + nB) * ldb;

    v8f acc = {};
#pragma unroll 4
    for (int kk = 0; kk < KD; kk += 32) {
        v16h a = load_a_frag16(As, KD, kk, lane);
        v16h b = load_b_frag16(Brow, kk, lane);
        acc = __builtin_amdgcn_wmma_f32_16x16x32_f16(false, a, false, b,
                                                     (short)0, acc, false, false);
    }
    const int   n  = n0 + nB;
    const int   mb = (lane < 16) ? 0 : 8;
    const float bv = bias ? bias[n] : 0.0f;
#pragma unroll
    for (int v = 0; v < 8; ++v) {
        const float r = acc[v] + bv;
        C[(size_t)(m0 + mb + v) * ldc + n] = r;
        if (C16) C16[(size_t)(m0 + mb + v) * ldc + n] = (_Float16)r;
    }
}

// ---- fused LSTM step: gates = x_t@W_ih^T + h@W_hh^T + b, then cell ---------
// grid = (B/16, H/16), block = 128; wave g owns gate g's 16x16 tile.
__global__ __launch_bounds__(128) void lstm_step_wmma(
    const _Float16* __restrict__ x16, int t,
    const _Float16* __restrict__ Wih16, const _Float16* __restrict__ Whh16,
    const float* __restrict__ b_ih, const float* __restrict__ b_hh,
    const _Float16* __restrict__ h_in, _Float16* __restrict__ h_out,
    float* __restrict__ c)
{
    __shared__ __align__(32) _Float16 xA[16 * I_DIM];   //  8 KB
    __shared__ __align__(32) _Float16 hA[16 * H_DIM];   // 16 KB
    __shared__ float gt[4 * 16 * 16];                   //  4 KB
    const int tid = threadIdx.x;
    const int b0 = blockIdx.x * 16, j0 = blockIdx.y * 16;

    stage_rows_async(xA, x16 + ((size_t)b0 * SDEC + t) * I_DIM, I_DIM,
                     (size_t)SDEC * I_DIM, tid);
    stage_rows_async(hA, h_in + (size_t)b0 * H_DIM, H_DIM, H_DIM, tid);
    wait_async_and_barrier();

    const int g = tid >> 5, lane = tid & 31;
    const int nB  = lane & 15;
    const int row = g * H_DIM + j0 + nB;               // W_ih / W_hh row (gate g)

    v8f acc = {};
    const _Float16* Wi = Wih16 + (size_t)row * I_DIM;
#pragma unroll 4
    for (int kk = 0; kk < I_DIM; kk += 32) {
        v16h a = load_a_frag16(xA, I_DIM, kk, lane);
        v16h b = load_b_frag16(Wi, kk, lane);
        acc = __builtin_amdgcn_wmma_f32_16x16x32_f16(false, a, false, b,
                                                     (short)0, acc, false, false);
    }
    const _Float16* Wh = Whh16 + (size_t)row * H_DIM;
#pragma unroll 4
    for (int kk = 0; kk < H_DIM; kk += 32) {
        v16h a = load_a_frag16(hA, H_DIM, kk, lane);
        v16h b = load_b_frag16(Wh, kk, lane);
        acc = __builtin_amdgcn_wmma_f32_16x16x32_f16(false, a, false, b,
                                                     (short)0, acc, false, false);
    }
    const float bsum = b_ih[row] + b_hh[row];
    const int   mb   = (lane < 16) ? 0 : 8;
#pragma unroll
    for (int v = 0; v < 8; ++v)
        gt[g * 256 + (mb + v) * 16 + nB] = acc[v] + bsum;
    __syncthreads();

    for (int p = tid; p < 256; p += 128) {
        const int m = p >> 4, n = p & 15;
        const float ig = sigmoidf_(gt[0 * 256 + p]);
        const float fg = sigmoidf_(gt[1 * 256 + p]);
        const float gg = tanhf    (gt[2 * 256 + p]);
        const float og = sigmoidf_(gt[3 * 256 + p]);
        const size_t idx = (size_t)(b0 + m) * H_DIM + (j0 + n);
        const float cn = fg * c[idx] + ig * gg;
        c[idx]     = cn;
        h_out[idx] = (_Float16)(og * tanhf(cn));
    }
}

// ---- per-batch-row attention + output head (K/V L2-resident) ---------------
__global__ __launch_bounds__(256) void attn_out_kernel(
    const _Float16* __restrict__ h16, const float* __restrict__ q,
    const float* __restrict__ Kb_, const float* __restrict__ Vb_,
    const float* __restrict__ Wfc, const float* __restrict__ bfc,
    float* __restrict__ out, int t)
{
    __shared__ __align__(16) float q_s[H_DIM];
    __shared__ float red[SENC];
    __shared__ float attn_s[SENC];
    __shared__ __align__(16) float hc[H_DIM];
    const int b = blockIdx.x, tid = threadIdx.x;

    __builtin_prefetch(Wfc + (size_t)tid * H_DIM, 0, 3);   // warm fc row in L2

    q_s[tid]       = q[(size_t)b * H_DIM + tid];
    q_s[256 + tid] = q[(size_t)b * H_DIM + 256 + tid];
    __syncthreads();

    const float4* Krow = (const float4*)(Kb_ + ((size_t)b * SENC + tid) * H_DIM);
    const float4* q4   = (const float4*)q_s;
    float dot = 0.0f;
#pragma unroll 4
    for (int i = 0; i < H_DIM / 4; ++i) {
        const float4 kv = Krow[i], qv = q4[i];
        dot += kv.x * qv.x + kv.y * qv.y + kv.z * qv.z + kv.w * qv.w;
    }
    const float sc = dot * 0.044194173824159216f;   // 1/sqrt(512)

    red[tid] = sc; __syncthreads();
    for (int off = 128; off > 0; off >>= 1) {
        if (tid < off) red[tid] = fmaxf(red[tid], red[tid + off]);
        __syncthreads();
    }
    const float mx = red[0]; __syncthreads();
    const float ex = __expf(sc - mx);
    red[tid] = ex; __syncthreads();
    for (int off = 128; off > 0; off >>= 1) {
        if (tid < off) red[tid] += red[tid + off];
        __syncthreads();
    }
    attn_s[tid] = ex * (1.0f / red[0]);
    __syncthreads();

    const float* Vb = Vb_ + (size_t)b * SENC * H_DIM;
    float c0 = 0.0f, c1 = 0.0f;
    for (int s = 0; s < SENC; ++s) {
        const float a = attn_s[s];
        c0 += a * Vb[(size_t)s * H_DIM + tid];
        c1 += a * Vb[(size_t)s * H_DIM + 256 + tid];
    }
    hc[tid]       = (float)h16[(size_t)b * H_DIM + tid] + c0;
    hc[256 + tid] = (float)h16[(size_t)b * H_DIM + 256 + tid] + c1;
    __syncthreads();

    const float4* Wr = (const float4*)(Wfc + (size_t)tid * H_DIM);
    const float4* h4 = (const float4*)hc;
    float acc = bfc[tid];
#pragma unroll 4
    for (int i = 0; i < H_DIM / 4; ++i) {
        const float4 wv = Wr[i], hv = h4[i];
        acc += wv.x * hv.x + wv.y * hv.y + wv.z * hv.z + wv.w * hv.w;
    }
    out[((size_t)b * SDEC + t) * O_DIM + tid] = acc;
}

__global__ void cvt_f32_f16_kernel(const float* __restrict__ src,
                                   _Float16* __restrict__ dst, int n) {
    const int i = blockIdx.x * blockDim.x + threadIdx.x;
    if (i < n) dst[i] = (_Float16)src[i];
}
__global__ void zero_u32_kernel(uint32_t* __restrict__ p, int n) {
    const int i = blockIdx.x * blockDim.x + threadIdx.x;
    if (i < n) p[i] = 0u;
}

extern "C" void kernel_launch(void* const* d_in, const int* in_sizes, int n_in,
                              void* d_out, int out_size, void* d_ws, size_t ws_size,
                              hipStream_t stream)
{
    (void)in_sizes; (void)n_in; (void)out_size; (void)ws_size;
    const float* x    = (const float*)d_in[0];
    const float* enc  = (const float*)d_in[1];
    const float* W_ih = (const float*)d_in[2];
    const float* W_hh = (const float*)d_in[3];
    const float* b_ih = (const float*)d_in[4];
    const float* b_hh = (const float*)d_in[5];
    const float* Wq   = (const float*)d_in[6];
    const float* bq   = (const float*)d_in[7];
    const float* Wk   = (const float*)d_in[8];
    const float* bk   = (const float*)d_in[9];
    const float* Wv   = (const float*)d_in[10];
    const float* bv   = (const float*)d_in[11];
    const float* Wfc  = (const float*)d_in[12];
    const float* bfc  = (const float*)d_in[13];
    float* out = (float*)d_out;

    // ---- workspace carve-up (256B-aligned), total ~224 MB ----
    char*  base = (char*)d_ws;
    size_t off  = 0;
    auto alloc = [&](size_t bytes) -> void* {
        void* p = base + off;
        off = (off + bytes + 255) & ~(size_t)255;
        return p;
    };
    const size_t KV = (size_t)B_DIM * SENC * H_DIM;           // 16,777,216
    float*     Kbuf  = (float*)    alloc(KV * 4);
    float*     Vbuf  = (float*)    alloc(KV * 4);
    float*     qb    = (float*)    alloc((size_t)B_DIM * H_DIM * 4);
    float*     cb    = (float*)    alloc((size_t)B_DIM * H_DIM * 4);
    _Float16*  h16a  = (_Float16*) alloc((size_t)B_DIM * H_DIM * 2);
    _Float16*  h16b  = (_Float16*) alloc((size_t)B_DIM * H_DIM * 2);
    _Float16*  x16   = (_Float16*) alloc((size_t)B_DIM * SDEC * I_DIM * 2);
    _Float16*  enc16 = (_Float16*) alloc(KV * 2);
    _Float16*  K16   = (_Float16*) alloc(KV * 2);
    _Float16*  Wih16 = (_Float16*) alloc((size_t)4 * H_DIM * I_DIM * 2);
    _Float16*  Whh16 = (_Float16*) alloc((size_t)4 * H_DIM * H_DIM * 2);
    _Float16*  Wq16  = (_Float16*) alloc((size_t)H_DIM * H_DIM * 2);
    _Float16*  Wk16  = (_Float16*) alloc((size_t)H_DIM * H_DIM * 2);
    _Float16*  Wv16  = (_Float16*) alloc((size_t)H_DIM * H_DIM * 2);

    const dim3 blk(128);
    auto cvt = [&](const float* s, _Float16* d, size_t n) {
        cvt_f32_f16_kernel<<<(n + 255) / 256, 256, 0, stream>>>(s, d, (int)n);
    };

    // one-time f32 -> f16 conversions (loop-invariant operands)
    cvt(W_ih, Wih16, (size_t)4 * H_DIM * I_DIM);
    cvt(W_hh, Whh16, (size_t)4 * H_DIM * H_DIM);
    cvt(Wq,   Wq16,  (size_t)H_DIM * H_DIM);
    cvt(Wk,   Wk16,  (size_t)H_DIM * H_DIM);
    cvt(Wv,   Wv16,  (size_t)H_DIM * H_DIM);
    cvt(x,    x16,   (size_t)B_DIM * SDEC * I_DIM);
    cvt(enc,  enc16, KV);

    // zero c (fp32) and initial h (f16)
    zero_u32_kernel<<<(B_DIM * H_DIM + 255) / 256, 256, 0, stream>>>(
        (uint32_t*)cb, B_DIM * H_DIM);
    zero_u32_kernel<<<(B_DIM * H_DIM / 2 + 255) / 256, 256, 0, stream>>>(
        (uint32_t*)h16a, B_DIM * H_DIM / 2);

    // hoisted encoder projections: K = enc@Wk^T + bk (fp32 + f16), V = K@Wv^T + bv
    gemm_bias_wmma_f16<<<dim3((B_DIM * SENC) / 16, H_DIM / 64), blk, 0, stream>>>(
        enc16, H_DIM, Wk16, H_DIM, bk, Kbuf, H_DIM, K16);
    gemm_bias_wmma_f16<<<dim3((B_DIM * SENC) / 16, H_DIM / 64), blk, 0, stream>>>(
        K16, H_DIM, Wv16, H_DIM, bv, Vbuf, H_DIM, nullptr);

    for (int t = 0; t < SDEC; ++t) {
        const _Float16* hin  = (t & 1) ? h16b : h16a;
        _Float16*       hout = (t & 1) ? h16a : h16b;   // double-buffered h
        lstm_step_wmma<<<dim3(B_DIM / 16, H_DIM / 16), blk, 0, stream>>>(
            x16, t, Wih16, Whh16, b_ih, b_hh, hin, hout, cb);
        gemm_bias_wmma_f16<<<dim3(B_DIM / 16, H_DIM / 64), blk, 0, stream>>>(
            hout, H_DIM, Wq16, H_DIM, bq, qb, H_DIM, nullptr);
        attn_out_kernel<<<dim3(B_DIM), dim3(256), 0, stream>>>(
            hout, qb, Kbuf, Vbuf, Wfc, bfc, out, t);
    }
}
